// SpatialAttentionBlock_10161892623062
// MI455X (gfx1250) — matrix operand order repaired and checked by reading, not compiled, not run on hardware
//
#include <hip/hip_runtime.h>

// ---------------------------------------------------------------------------
// SpatialAttentionBlock for MI455X (gfx1250): f16 WMMA w/ f32 accumulation.
// b=16, c=2048, h=w=32 (hw=1024), d_att=d_emb=64.
// ---------------------------------------------------------------------------

typedef _Float16 h8   __attribute__((ext_vector_type(8)));
typedef _Float16 h16  __attribute__((ext_vector_type(16)));
typedef float    f8v  __attribute__((ext_vector_type(8)));
typedef float    f4v  __attribute__((ext_vector_type(4)));

static constexpr int kNB = 16;    // batch
static constexpr int kC  = 2048;  // channels
static constexpr int kHW = 1024;  // h*w
static constexpr int kD  = 64;    // atten/emb dim

union FragU { h16 v; struct { h8 lo, hi; } s; };

// ---------------------------------------------------------------------------
// Kernel 0: convert wq|wk|wv (each [64,2048] f32) into one f16 buffer
// wh layout: [3][64][2048], p=0:q, 1:k, 2:v
// ---------------------------------------------------------------------------
__global__ __launch_bounds__(256)
void cvt_weights_kernel(const float* __restrict__ wq,
                        const float* __restrict__ wk,
                        const float* __restrict__ wv,
                        _Float16* __restrict__ wh) {
  const int n = kD * kC;
  int i = blockIdx.x * 256 + threadIdx.x;
  float v;
  if (i < n)            v = wq[i];
  else if (i < 2 * n)   v = wk[i - n];
  else                  v = wv[i - 2 * n];
  wh[i] = (_Float16)v;
}

// ---------------------------------------------------------------------------
// Kernel 1: fused Q/K/V projection.
//   q[d,n] = sum_c Wq[d,c] x[c,n]  (+bias), same for k, v.
// Grid: (hw/128, b), 256 threads (8 waves). Wave w owns columns [w*16, w*16+16).
// A-fragments (weights) from global f16 (K-contiguous).
// B-fragments from LDS-staged x tile (transposed to [n][k], f16).
// Outputs: qt/kt = [b, hw, 64] (transposed, f16), vv = [b, 64, hw] (f16).
// ---------------------------------------------------------------------------
__global__ __launch_bounds__(256)
void proj_kernel(const float* __restrict__ x,
                 const _Float16* __restrict__ wh,
                 const float* __restrict__ bq,
                 const float* __restrict__ bk,
                 const float* __restrict__ bv,
                 _Float16* __restrict__ qt,
                 _Float16* __restrict__ kt,
                 _Float16* __restrict__ vv) {
  __shared__ _Float16 Xs[128][40];  // [n][k], padded row = 80B (16B aligned)

  const int b    = blockIdx.y;
  const int n0   = blockIdx.x * 128;
  const int t    = threadIdx.x;
  const int lane = t & 31;
  const int wave = t >> 5;          // 0..7
  const int lm   = lane & 15;       // row/col within 16
  const int lk   = lane >> 4;       // K-half selector

  const float* xb = x + (size_t)b * kC * kHW;

  f8v acc[3][4] = {};               // [proj][mtile] accumulators, f32 16x16

  // staging role for this thread
  const int sn = t & 127;           // column in tile
  const int sk = t >> 7;            // 0/1 -> K rows [sk*16, sk*16+16)

  for (int c0 = 0; c0 < kC; c0 += 32) {
    __syncthreads();
    // ---- stage 32(K) x 128(N) tile of x into LDS, transposed, f16 ----
    {
      const float* src = xb + (size_t)(c0 + sk * 16) * kHW + n0 + sn;
      __builtin_prefetch(src + 32 * kHW, 0, 1);  // next K-block
#pragma unroll
      for (int i = 0; i < 16; ++i) {
        Xs[sn][sk * 16 + i] = (_Float16)src[i * kHW];
      }
    }
    __syncthreads();

    // ---- B fragment: column n = wave*16+lm, K runs at lk*8 and 16+lk*8 ----
    FragU bf;
    {
      const _Float16* p = &Xs[wave * 16 + lm][lk * 8];
      bf.s.lo = *(const h8*)p;
      bf.s.hi = *(const h8*)(p + 16);
    }

    // ---- A fragments (weights) + WMMA, 3 projections x 4 M-tiles ----
#pragma unroll
    for (int p = 0; p < 3; ++p) {
      const _Float16* wbase = wh + (size_t)p * kD * kC;
#pragma unroll
      for (int mt = 0; mt < 4; ++mt) {
        const _Float16* wr =
            wbase + (size_t)(mt * 16 + lm) * kC + c0 + lk * 8;
        FragU af;
        af.s.lo = *(const h8*)wr;         // K run [lk*8, lk*8+8)
        af.s.hi = *(const h8*)(wr + 16);  // K run [16+lk*8, ...)
        acc[p][mt] = __builtin_amdgcn_wmma_f32_16x16x32_f16(
            false, af.v, false, bf.v, (short)0, acc[p][mt], false, false);
      }
    }
  }

  // ---- epilogue: bias add + store ----
  const int ncol = n0 + wave * 16 + lm;   // this lane's output column
#pragma unroll
  for (int mt = 0; mt < 4; ++mt) {
    const int dbase = mt * 16 + lk * 8;   // 8 consecutive d per lane
    const f4v q0 = *(const f4v*)(bq + dbase);
    const f4v q1 = *(const f4v*)(bq + dbase + 4);
    const f4v k0 = *(const f4v*)(bk + dbase);
    const f4v k1 = *(const f4v*)(bk + dbase + 4);
    const f4v v0 = *(const f4v*)(bv + dbase);
    const f4v v1 = *(const f4v*)(bv + dbase + 4);

    h8 hq, hk;
#pragma unroll
    for (int r = 0; r < 8; ++r) {
      float bqr = (r < 4) ? q0[r] : q1[r - 4];
      float bkr = (r < 4) ? k0[r] : k1[r - 4];
      hq[r] = (_Float16)(acc[0][mt][r] + bqr);
      hk[r] = (_Float16)(acc[1][mt][r] + bkr);
    }
    // qt/kt: [b, n, d] -> 8 contiguous halves (16B) per lane
    *(h8*)(qt + ((size_t)b * kHW + ncol) * kD + dbase) = hq;
    *(h8*)(kt + ((size_t)b * kHW + ncol) * kD + dbase) = hk;
    // vv: [b, d, n] -> strided b16 stores
#pragma unroll
    for (int r = 0; r < 8; ++r) {
      float bvr = (r < 4) ? v0[r] : v1[r - 4];
      vv[((size_t)b * kD + dbase + r) * kHW + ncol] =
          (_Float16)(acc[2][mt][r] + bvr);
    }
  }
}

// ---------------------------------------------------------------------------
// Kernel 2: attention for one batch x 16 query rows.
//   S = (Q^T K)/64, P = exp(S - rowmax), O = (P V^T) / rowsum.
// Grid: (hw/16, b), 128 threads (4 waves).
// ---------------------------------------------------------------------------
__global__ __launch_bounds__(128)
void attn_kernel(const _Float16* __restrict__ qt,
                 const _Float16* __restrict__ kt,
                 const _Float16* __restrict__ vv,
                 float* __restrict__ out) {
  __shared__ _Float16 SP[16][kHW];  // scores -> probabilities, 32KB
  __shared__ float red[16][8];
  __shared__ float rowmax[16];
  __shared__ float rinv[16];

  const int b    = blockIdx.y;
  const int x0   = blockIdx.x * 16;
  const int t    = threadIdx.x;
  const int lane = t & 31;
  const int wave = t >> 5;      // 0..3
  const int lm   = lane & 15;
  const int lk   = lane >> 4;

  // ---- hoist Q A-fragments: M=16 query rows, K=d (two 32-wide k-steps) ----
  const _Float16* qrow = qt + ((size_t)b * kHW + x0 + lm) * kD;
  FragU qa[2];
#pragma unroll
  for (int s = 0; s < 2; ++s) {
    const _Float16* p = qrow + s * 32 + lk * 8;
    qa[s].s.lo = *(const h8*)p;
    qa[s].s.hi = *(const h8*)(p + 16);
  }

  // ---- phase 1: S tiles (each wave handles 16 of 64 j-tiles) ----
  for (int jt = wave; jt < 64; jt += 4) {
    const int j0 = jt * 16;
    f8v s = {};
#pragma unroll
    for (int step = 0; step < 2; ++step) {
      // B fragment from kt: column y = j0+lm, d-runs at step*32+lk*8 and +16
      const _Float16* kp =
          kt + ((size_t)b * kHW + j0 + lm) * kD + step * 32 + lk * 8;
      FragU bf;
      bf.s.lo = *(const h8*)kp;
      bf.s.hi = *(const h8*)(kp + 16);
      s = __builtin_amdgcn_wmma_f32_16x16x32_f16(
          false, qa[step].v, false, bf.v, (short)0, s, false, false);
    }
    // store scaled scores: lane = column (j0+lm), VGPR r = row (lk*8+r)
#pragma unroll
    for (int r = 0; r < 8; ++r) {
      SP[lk * 8 + r][j0 + lm] = (_Float16)(s[r] * 0.015625f);  // 1/64
    }
  }
  __syncthreads();

  // ---- phase 2: row softmax (unnormalized P + reciprocal row-sum) ----
  {
    const int row = t & 15;
    const int seg = t >> 4;  // 0..7, each covers 128 columns
    float mx = -3.0e38f;
    for (int j = seg * 128; j < seg * 128 + 128; ++j) {
      mx = fmaxf(mx, (float)SP[row][j]);
    }
    red[row][seg] = mx;
    __syncthreads();
    if (seg == 0) {
      float m = red[row][0];
#pragma unroll
      for (int i = 1; i < 8; ++i) m = fmaxf(m, red[row][i]);
      rowmax[row] = m;
    }
    __syncthreads();
    const float m = rowmax[row];
    float sum = 0.0f;
    for (int j = seg * 128; j < seg * 128 + 128; ++j) {
      float p = __expf((float)SP[row][j] - m);
      sum += p;
      SP[row][j] = (_Float16)p;
    }
    red[row][seg] = sum;
    __syncthreads();
    if (seg == 0) {
      float ssum = red[row][0];
#pragma unroll
      for (int i = 1; i < 8; ++i) ssum += red[row][i];
      rinv[row] = 1.0f / ssum;
    }
    __syncthreads();
  }

  // ---- phase 3: O = P V^T, wave owns d-tile [wave*16, wave*16+16) ----
  {
    const int d0 = wave * 16;
    f8v o = {};
    for (int j0 = 0; j0 < kHW; j0 += 32) {
      // A fragment from LDS P: row x=lm, K(j) runs at j0+lk*8 and j0+16+lk*8
      const _Float16* pp = &SP[lm][j0 + lk * 8];
      FragU af;
      af.s.lo = *(const h8*)pp;
      af.s.hi = *(const h8*)(pp + 16);
      // B fragment from vv: column d = d0+lm, K(j) runs at j0+lk*8 and j0+16+lk*8
      const _Float16* vp =
          vv + ((size_t)b * kD + d0 + lm) * kHW + j0 + lk * 8;
      FragU bf;
      bf.s.lo = *(const h8*)vp;
      bf.s.hi = *(const h8*)(vp + 16);
      o = __builtin_amdgcn_wmma_f32_16x16x32_f16(
          false, af.v, false, bf.v, (short)0, o, false, false);
    }
    // store: lane = column d, VGPR r = row x (contiguous in out's hw dim)
    float* op = out + ((size_t)b * kD + d0 + lm) * kHW + x0 + lk * 8;
    float vals[8];
#pragma unroll
    for (int r = 0; r < 8; ++r) vals[r] = o[r] * rinv[lk * 8 + r];
    *(f4v*)op       = f4v{vals[0], vals[1], vals[2], vals[3]};
    *(f4v*)(op + 4) = f4v{vals[4], vals[5], vals[6], vals[7]};
  }
}

// ---------------------------------------------------------------------------
extern "C" void kernel_launch(void* const* d_in, const int* in_sizes, int n_in,
                              void* d_out, int out_size, void* d_ws,
                              size_t ws_size, hipStream_t stream) {
  const float* x  = (const float*)d_in[0];
  const float* wq = (const float*)d_in[1];
  const float* bq = (const float*)d_in[2];
  const float* wk = (const float*)d_in[3];
  const float* bk = (const float*)d_in[4];
  const float* wv = (const float*)d_in[5];
  const float* bv = (const float*)d_in[6];
  float* out = (float*)d_out;

  char* ws = (char*)d_ws;
  size_t off = 0;
  _Float16* wh = (_Float16*)(ws + off);
  off += (size_t)3 * kD * kC * sizeof(_Float16);          // 768 KB
  off = (off + 255) & ~(size_t)255;
  _Float16* qt = (_Float16*)(ws + off);
  off += (size_t)kNB * kHW * kD * sizeof(_Float16);       // 2 MB
  _Float16* kt = (_Float16*)(ws + off);
  off += (size_t)kNB * kHW * kD * sizeof(_Float16);       // 2 MB
  _Float16* vv = (_Float16*)(ws + off);                   // 2 MB

  cvt_weights_kernel<<<dim3((3 * kD * kC) / 256), 256, 0, stream>>>(
      wq, wk, wv, wh);
  proj_kernel<<<dim3(kHW / 128, kNB), 256, 0, stream>>>(
      x, wh, bq, bk, bv, qt, kt, vv);
  attn_kernel<<<dim3(kHW / 16, kNB), 128, 0, stream>>>(qt, kt, vv, out);
}
